// MultiHeadSelfAttention2d_1743756722341
// MI455X (gfx1250) — matrix operand order repaired and checked by reading, not compiled, not run on hardware
//
#include <hip/hip_runtime.h>
#include <cmath>

// ---------------------------------------------------------------------------
// MultiHeadSelfAttention2d for MI455X (gfx1250, wave32, WMMA f16 16x16x32)
//   B=1, H=12, C=32, N=S*L=4096, D=384, QKV rows=1152
//   d_out = [ out: 384*4096 f32 | attn: 12*4096*4096 f32 ]
// ---------------------------------------------------------------------------

typedef __attribute__((ext_vector_type(16))) _Float16 v16h;
typedef __attribute__((ext_vector_type(8)))  _Float16 v8h;
typedef __attribute__((ext_vector_type(8)))  float    v8f;

#define HEADS   12
#define CDIM    32
#define NTOK    4096
#define DMODEL  384
#define QKVROWS 1152
#define QSCALE  0.17677669529663687f   // 1/sqrt(32)

// A-operand (16x32 f16): lane<16 -> M=lane, K in {0..15}; lane>=16 -> M=lane-16,
// K in {16..31}. Two contiguous 8-half (16B) chunks per lane.
__device__ __forceinline__ v16h load_a_f16(const _Float16* p0, const _Float16* p1) {
  union { v16h v; v8h h[2]; } u;
  u.h[0] = *reinterpret_cast<const v8h*>(p0);
  u.h[1] = *reinterpret_cast<const v8h*>(p1);
  return u.v;
}
// B-operand (32x16 f16): lane<16 -> N=lane, K=0..15; lane>=16 -> N=lane-16, K=16..31.
// One contiguous 16-half (32B) run per lane.
__device__ __forceinline__ v16h load_b_f16(const _Float16* p) {
  union { v16h v; v8h h[2]; } u;
  u.h[0] = *reinterpret_cast<const v8h*>(p);
  u.h[1] = *reinterpret_cast<const v8h*>(p + 8);
  return u.v;
}

// ---------------------------------------------------------------------------
// Kernel 1: fp32 -> f16 conversions.  x[384][4096] -> xT[4096][384] (token-major),
// W[1152][384] -> wsW (same layout).
// ---------------------------------------------------------------------------
__global__ void cvt_kernel(const float* __restrict__ x, const float* __restrict__ W,
                           _Float16* __restrict__ wsW, _Float16* __restrict__ xT) {
  int idx = blockIdx.x * blockDim.x + threadIdx.x;
  const int NX = DMODEL * NTOK;       // 1,572,864
  const int NW = QKVROWS * DMODEL;    // 442,368
  if (idx < NX) {
    int k = idx >> 12;                // channel
    int n = idx & (NTOK - 1);         // token
    xT[n * DMODEL + k] = (_Float16)x[idx];
  } else if (idx < NX + NW) {
    int i = idx - NX;
    wsW[i] = (_Float16)W[i];
  }
}

// ---------------------------------------------------------------------------
// Kernel 2: QKV projection  qkv[1152][4096] = W @ x + b   (K=384 -> 12 WMMAs/tile)
// q (scaled) -> qT[h][n][32], k -> kT[h][n][32], v -> vC[h][c][4096]  (all f16)
// One wave per 16x16 output tile. grid = (256 token tiles, 72 row tiles)
// ---------------------------------------------------------------------------
__global__ void qkv_kernel(const _Float16* __restrict__ wsW,
                           const _Float16* __restrict__ xT,
                           const float*    __restrict__ bias,
                           _Float16* __restrict__ qT,
                           _Float16* __restrict__ kT,
                           _Float16* __restrict__ vC) {
  const int lane = threadIdx.x;
  const int col  = lane & 15;
  const int hi   = lane >> 4;
  const int base = hi * 8;
  const int n0   = blockIdx.x * 16;   // token tile
  const int m0   = blockIdx.y * 16;   // qkv-row tile

  const _Float16* arow = wsW + (size_t)(m0 + col) * DMODEL;  // A row (W)
  const _Float16* brow = xT  + (size_t)(n0 + col) * DMODEL;  // B col (token)

  v8f acc = {};
#pragma unroll
  for (int kb = 0; kb < DMODEL; kb += 32) {
    v16h a = load_a_f16(arow + kb + hi * 16, arow + kb + hi * 16 + 8);
    v16h b = load_b_f16(brow + kb + hi * 16);
    acc = __builtin_amdgcn_wmma_f32_16x16x32_f16(false, a, false, b,
                                                 (short)0, acc, false, false);
  }

  // Bias: D element r maps to row m = m0 + r + 8*hi (contiguous in r).
  const float4* bp = reinterpret_cast<const float4*>(bias + m0 + 8 * hi);
  float4 bv0 = bp[0], bv1 = bp[1];
  float bb[8] = {bv0.x, bv0.y, bv0.z, bv0.w, bv1.x, bv1.y, bv1.z, bv1.w};

  const int n = n0 + col;             // this lane's token
  if (m0 < DMODEL) {                  // ---- q (scaled) ----
    int h  = m0 >> 5;
    int c0 = (m0 & 31) + 8 * hi;      // c = c0 + r, contiguous
    v8h o;
#pragma unroll
    for (int r = 0; r < 8; ++r) o[r] = (_Float16)((acc[r] + bb[r]) * QSCALE);
    *reinterpret_cast<v8h*>(qT + ((size_t)(h * NTOK + n) * CDIM) + c0) = o;
  } else if (m0 < 2 * DMODEL) {       // ---- k ----
    int mk = m0 - DMODEL;
    int h  = mk >> 5;
    int c0 = (mk & 31) + 8 * hi;
    v8h o;
#pragma unroll
    for (int r = 0; r < 8; ++r) o[r] = (_Float16)(acc[r] + bb[r]);
    *reinterpret_cast<v8h*>(kT + ((size_t)(h * NTOK + n) * CDIM) + c0) = o;
  } else {                            // ---- v (channel-major) ----
    int mv = m0 - 2 * DMODEL;
    int h  = mv >> 5;
    int c0 = (mv & 31) + 8 * hi;
#pragma unroll
    for (int r = 0; r < 8; ++r)
      vC[(size_t)(h * CDIM + c0 + r) * NTOK + n] = (_Float16)(acc[r] + bb[r]);
  }
}

// ---------------------------------------------------------------------------
// Kernel 3: raw attention scores  S[h][i][j] = sum_c qT[h][i][c] * kT[h][j][c]
// K=C=32 -> exactly ONE v_wmma_f32_16x16x32_f16 per 16x16 tile.
// LDS transpose -> coalesced 32B stores. grid = (256 i-tiles, 16 j-groups, 12 heads)
// ---------------------------------------------------------------------------
__global__ void scores_kernel(const _Float16* __restrict__ qT,
                              const _Float16* __restrict__ kT,
                              float* __restrict__ attn) {
  __shared__ float tile[16 * 16];
  const int lane = threadIdx.x;
  const int col  = lane & 15;
  const int hi   = lane >> 4;
  const int base = hi * 8;
  const int i0   = blockIdx.x * 16;
  const int h    = blockIdx.z;

  const _Float16* arow = qT + (size_t)(h * NTOK + i0 + col) * CDIM;
  v16h a = load_a_f16(arow + hi * 16, arow + hi * 16 + 8);

  const int ro = lane >> 1, hf = lane & 1;   // for coalesced tile store
  for (int t = 0; t < 16; ++t) {
    const int j0 = (blockIdx.y * 16 + t) * 16;
    const _Float16* brow = kT + (size_t)(h * NTOK + j0 + col) * CDIM + hi * 16;
    v16h b = load_b_f16(brow);
    v8f c = {};
    c = __builtin_amdgcn_wmma_f32_16x16x32_f16(false, a, false, b,
                                               (short)0, c, false, false);
    // D element: (M = r + 8*hi, N = col) -> transpose through LDS
#pragma unroll
    for (int r = 0; r < 8; ++r) tile[(r + 8 * hi) * 16 + col] = c[r];
    __syncthreads();
    const float4* src = reinterpret_cast<const float4*>(&tile[ro * 16 + hf * 8]);
    float4* dst = reinterpret_cast<float4*>(
        attn + (size_t)(h * NTOK + i0 + ro) * NTOK + j0 + hf * 8);
    dst[0] = src[0];
    dst[1] = src[1];
    __syncthreads();
  }
}

// ---------------------------------------------------------------------------
// Kernel 4: per-row softmax stats (max, 1/sum(exp)).  One 256-thread block per row.
// ---------------------------------------------------------------------------
__global__ void stats_kernel(const float* __restrict__ attn,
                             float* __restrict__ rowmax,
                             float* __restrict__ rowinv) {
  __shared__ float red[256];
  const int row = blockIdx.y * NTOK + blockIdx.x;
  const int t   = threadIdx.x;
  const float4* p = reinterpret_cast<const float4*>(attn + (size_t)row * NTOK);

  float4 v[4];
  float mx = -INFINITY;
#pragma unroll
  for (int c = 0; c < 4; ++c) {
    v[c] = p[t + c * 256];
    mx = fmaxf(mx, fmaxf(fmaxf(v[c].x, v[c].y), fmaxf(v[c].z, v[c].w)));
  }
  red[t] = mx;
  __syncthreads();
  for (int s = 128; s > 0; s >>= 1) {
    if (t < s) red[t] = fmaxf(red[t], red[t + s]);
    __syncthreads();
  }
  const float m = red[0];
  __syncthreads();

  float sum = 0.f;
#pragma unroll
  for (int c = 0; c < 4; ++c) {
    sum += __expf(v[c].x - m) + __expf(v[c].y - m) +
           __expf(v[c].z - m) + __expf(v[c].w - m);
  }
  red[t] = sum;
  __syncthreads();
  for (int s = 128; s > 0; s >>= 1) {
    if (t < s) red[t] += red[t + s];
    __syncthreads();
  }
  if (t == 0) {
    rowmax[row] = m;
    rowinv[row] = 1.f / red[0];
  }
}

// ---------------------------------------------------------------------------
// Kernel 5: fused exp-normalize (writes final attn) + out = attn @ v^T.
// One wave per (head, 16-row block); 128 j-steps, 2 WMMAs each (c-blocks 0,1).
// A operand is built in-registers from the just-computed attn values.
// ---------------------------------------------------------------------------
__global__ void out_kernel(const _Float16* __restrict__ vC,
                           const float* __restrict__ rowmax,
                           const float* __restrict__ rowinv,
                           float* __restrict__ attn,
                           float* __restrict__ out) {
  const int lane = threadIdx.x;
  const int col  = lane & 15;
  const int hi   = lane >> 4;
  const int base = hi * 8;
  const int i0   = blockIdx.x * 16;
  const int h    = blockIdx.y;

  const int row = h * NTOK + i0 + col;            // this lane's A row
  const float m = rowmax[row];
  const float s = rowinv[row];
  float* arow = attn + (size_t)row * NTOK;

  const _Float16* vrow0 = vC + (size_t)(h * CDIM + col)      * NTOK;  // c-block 0
  const _Float16* vrow1 = vC + (size_t)(h * CDIM + 16 + col) * NTOK;  // c-block 1

  v8f acc0 = {}, acc1 = {};
  for (int jb = 0; jb < NTOK; jb += 32) {
    float4* p0 = reinterpret_cast<float4*>(arow + jb + hi * 16);     // K {16*hi..+7}
    float4* p1 = reinterpret_cast<float4*>(arow + jb + hi * 16 + 8); // K {16*hi+8..+15}
    float raw[16];
    *reinterpret_cast<float4*>(&raw[0])  = p0[0];
    *reinterpret_cast<float4*>(&raw[4])  = p0[1];
    *reinterpret_cast<float4*>(&raw[8])  = p1[0];
    *reinterpret_cast<float4*>(&raw[12]) = p1[1];
#pragma unroll
    for (int t = 0; t < 16; ++t) raw[t] = __expf(raw[t] - m) * s;
    // write final normalized attention
    p0[0] = *reinterpret_cast<float4*>(&raw[0]);
    p0[1] = *reinterpret_cast<float4*>(&raw[4]);
    p1[0] = *reinterpret_cast<float4*>(&raw[8]);
    p1[1] = *reinterpret_cast<float4*>(&raw[12]);
    // build A operand from the same values (contiguous k = 16*hi + t, same order as B)
    union { v16h v; v8h h2[2]; } ua;
#pragma unroll
    for (int t = 0; t < 8; ++t) {
      ua.h2[0][t] = (_Float16)raw[t];
      ua.h2[1][t] = (_Float16)raw[8 + t];
    }
    v16h b0 = load_b_f16(vrow0 + jb + hi * 16);
    v16h b1 = load_b_f16(vrow1 + jb + hi * 16);
    acc0 = __builtin_amdgcn_wmma_f32_16x16x32_f16(false, ua.v, false, b0,
                                                  (short)0, acc0, false, false);
    acc1 = __builtin_amdgcn_wmma_f32_16x16x32_f16(false, ua.v, false, b1,
                                                  (short)0, acc1, false, false);
    if (jb + 32 < NTOK) __builtin_prefetch(arow + jb + 32, 0, 1);
  }

  // D element: (i_local = r + 8*hi, c_local = col); out layout [h*32+c][n]
#pragma unroll
  for (int r = 0; r < 8; ++r) {
    const int i = i0 + r + 8 * hi;
    out[(size_t)(h * CDIM + col)      * NTOK + i] = acc0[r];
    out[(size_t)(h * CDIM + 16 + col) * NTOK + i] = acc1[r];
  }
}

// ---------------------------------------------------------------------------
extern "C" void kernel_launch(void* const* d_in, const int* in_sizes, int n_in,
                              void* d_out, int out_size, void* d_ws, size_t ws_size,
                              hipStream_t stream) {
  const float* x = (const float*)d_in[0];   // [384, 4096] fp32
  const float* W = (const float*)d_in[1];   // [1152, 384] fp32
  const float* b = (const float*)d_in[2];   // [1152] fp32

  float* out  = (float*)d_out;              // [384*4096]
  float* attn = out + DMODEL * NTOK;        // [12*4096*4096]

  char* ws = (char*)d_ws;                   // ~13.9 MB used, 256B-aligned regions
  _Float16* wsW    = (_Float16*)(ws + 0);          //  884,736 B
  _Float16* xT     = (_Float16*)(ws + 884736);     // 3,145,728 B
  _Float16* qT     = (_Float16*)(ws + 4030464);    // 3,145,728 B
  _Float16* kT     = (_Float16*)(ws + 7176192);    // 3,145,728 B
  _Float16* vC     = (_Float16*)(ws + 10321920);   // 3,145,728 B
  float*    rowmax = (float*)   (ws + 13467648);   //   196,608 B
  float*    rowinv = (float*)   (ws + 13664256);   //   196,608 B

  cvt_kernel   <<<7872, 256, 0, stream>>>(x, W, wsW, xT);
  qkv_kernel   <<<dim3(NTOK / 16, QKVROWS / 16), 32, 0, stream>>>(wsW, xT, b, qT, kT, vC);
  scores_kernel<<<dim3(NTOK / 16, 16, HEADS),    32, 0, stream>>>(qT, kT, attn);
  stats_kernel <<<dim3(NTOK, HEADS),            256, 0, stream>>>(attn, rowmax, rowinv);
  out_kernel   <<<dim3(NTOK / 16, HEADS),        32, 0, stream>>>(vC, rowmax, rowinv, attn, out);
}
